// LinearMoleLayer_4999341932691
// MI455X (gfx1250) — matrix operand order repaired and checked by reading, not compiled, not run on hardware
//
#include <hip/hip_runtime.h>
#include <math.h>

// ---------------------------------------------------------------------------
// MoE-LoRA fused layer for MI455X (gfx1250, wave32, WMMA).
//   out = x @ W_base.T + b_base + SCALING * sum_e w[n,e] * (x @ A[e].T) @ B[e].T
// LoRA up-proj folded into the base GEMM as 128 extra K columns (K = 4224).
// GEMMs: bf16 WMMA, double-buffered LDS staging fed by async global->LDS
// copies (ASYNCcnt), so loads of slab s+1 overlap WMMAs of slab s.
// KC = 64: two WMMA k-steps per stage -> half the barrier overhead per WMMA.
// ---------------------------------------------------------------------------

typedef __attribute__((ext_vector_type(16))) __bf16 v16bf;
typedef __attribute__((ext_vector_type(8)))  float  v8f;

#define MOE_SCALING 2.0f  // lora_alpha / lora_rank = 32/16

// ---------------- async global->LDS copy (gfx1250) --------------------------
// Signature (confirmed by round-2 diagnostic): (v4i32 AS(1)*, v4i32 AS(3)*,
// imm offset, imm cpol)
typedef __attribute__((__vector_size__(16))) int v4i;
typedef __attribute__((address_space(1))) v4i gv4i;
typedef __attribute__((address_space(3))) v4i lv4i;

#if __has_builtin(__builtin_amdgcn_global_load_async_to_lds_b128)
#define ASYNC_COPY 1
__device__ __forceinline__ void cp_b128(const __bf16* g, __bf16* l) {
  __builtin_amdgcn_global_load_async_to_lds_b128((gv4i*)g, (lv4i*)l, 0, 0);
}
#else
#define ASYNC_COPY 0
__device__ __forceinline__ void cp_b128(const __bf16* g, __bf16* l) {
  *(uint4*)l = *(const uint4*)g;  // sync fallback: global_load + ds_store
}
#endif

template <int N>
__device__ __forceinline__ void wait_async() {
#if ASYNC_COPY
#if __has_builtin(__builtin_amdgcn_s_wait_asynccnt)
  __builtin_amdgcn_s_wait_asynccnt(N);
#else
  asm volatile("s_wait_asynccnt %0" ::"n"(N) : "memory");
#endif
#endif
}

union frag16 {
  v16bf v;
  uint4 u[2];
};

// ---------------- LDS-staged bf16 WMMA GEMM ---------------------------------
// C[M, ldc] = A[M, lda] * B[N, ldb]^T (+ bias[n]); A/B bf16 row-major over K.
// Block tile (WAVES_M*WM*16) x (WAVES_N*WN*16); 32*WAVES_M*WAVES_N threads.
template <int WM, int WN, int WAVES_M, int WAVES_N, bool ADD_BIAS>
__global__ __launch_bounds__(WAVES_M* WAVES_N * 32)
void gemm_bf16_wmma_lds(const __bf16* __restrict__ Ag, int lda,
                        const __bf16* __restrict__ Bg, int ldb,
                        float* __restrict__ Cg, int ldc,
                        const float* __restrict__ bias, int K) {
  constexpr int THREADS = WAVES_M * WAVES_N * 32;
  constexpr int BM = WAVES_M * WM * 16;
  constexpr int BN = WAVES_N * WN * 16;
  constexpr int KC = 64;            // K elements per stage (2 WMMA k-steps)
  constexpr int LSTR = KC + 8;      // padded LDS row stride (72 elems = 144 B)
  constexpr int A_ITER = (BM * (KC / 8)) / THREADS;  // b128 chunks per thread
  constexpr int B_ITER = (BN * (KC / 8)) / THREADS;
  constexpr int STAGE_OPS = A_ITER + B_ITER;  // async instrs per wave / stage

  __shared__ __bf16 shA[2][BM * LSTR];
  __shared__ __bf16 shB[2][BN * LSTR];

  const int tid = threadIdx.x;
  const int wave = tid >> 5;
  const int lane = tid & 31;
  const int wm = wave % WAVES_M;
  const int wn = wave / WAVES_M;
  const int block_m = blockIdx.y * BM;
  const int block_n = blockIdx.x * BN;
  const int r = lane & 15;
  const int hi = lane >> 4;
  const int tmw = wm * WM * 16;
  const int tnw = wn * WN * 16;

  v8f acc[WM][WN];
#pragma unroll
  for (int i = 0; i < WM; ++i)
#pragma unroll
    for (int j = 0; j < WN; ++j) acc[i][j] = (v8f){};

  auto stage = [&](int buf, int k0) {
#pragma unroll
    for (int c = 0; c < A_ITER; ++c) {
      const int ch = tid + c * THREADS;
      const int row = ch >> 3;  // KC/8 = 8 chunks per row
      const int seg = ch & 7;
      cp_b128(Ag + (size_t)(block_m + row) * (size_t)lda + k0 + seg * 8,
              &shA[buf][row * LSTR + seg * 8]);
    }
#pragma unroll
    for (int c = 0; c < B_ITER; ++c) {
      const int ch = tid + c * THREADS;
      const int row = ch >> 3;
      const int seg = ch & 7;
      cp_b128(Bg + (size_t)(block_n + row) * (size_t)ldb + k0 + seg * 8,
              &shB[buf][row * LSTR + seg * 8]);
    }
  };

  const int nstages = K / KC;
  stage(0, 0);

  for (int s = 0; s < nstages; ++s) {
    const int buf = s & 1;
    if (s + 1 < nstages) {
      stage(buf ^ 1, (s + 1) * KC);
      wait_async<STAGE_OPS>();  // stage s done; s+1 still in flight
    } else {
      wait_async<0>();
    }
    __syncthreads();

#pragma unroll
    for (int kk = 0; kk < KC; kk += 32) {
      // A operand (16x32): lane -> row L&15, elems {kb..kb+7, kb+16..kb+23}.
      frag16 a[WM];
#pragma unroll
      for (int i = 0; i < WM; ++i) {
        const __bf16* p = &shA[buf][(tmw + i * 16 + r) * LSTR + kk + hi * 8];
        a[i].u[0] = *(const uint4*)p;
        a[i].u[1] = *(const uint4*)(p + 16);
      }
      // B operand (32x16): lane -> col L&15, elems {kb..kb+7, kb+16..kb+23}.
      frag16 b[WN];
#pragma unroll
      for (int j = 0; j < WN; ++j) {
        const __bf16* p = &shB[buf][(tnw + j * 16 + r) * LSTR + kk + hi * 8];
        b[j].u[0] = *(const uint4*)p;
        b[j].u[1] = *(const uint4*)(p + 16);
      }
#pragma unroll
      for (int i = 0; i < WM; ++i)
#pragma unroll
        for (int j = 0; j < WN; ++j)
          acc[i][j] = __builtin_amdgcn_wmma_f32_16x16x32_bf16(
              false, a[i].v, false, b[j].v, (short)0, acc[i][j], false, false);
    }

    __syncthreads();  // all waves done reading buf before it is restaged
  }

  // C/D layout: lane L -> col L&15, rows (L>>4)*8 + v, v = 0..7.
  const int rb = hi * 8;
#pragma unroll
  for (int i = 0; i < WM; ++i) {
#pragma unroll
    for (int j = 0; j < WN; ++j) {
      const int cn = block_n + tnw + j * 16 + r;
      const float bv = ADD_BIAS ? bias[cn] : 0.0f;
#pragma unroll
      for (int v = 0; v < 8; ++v) {
        const int rm = block_m + tmw + i * 16 + rb + v;
        Cg[(size_t)rm * (size_t)ldc + cn] = acc[i][j][v] + bv;
      }
    }
  }
}

// ---------------- small helper kernels --------------------------------------
__global__ void build_wb_kernel(const float* __restrict__ Wb,
                                const float* __restrict__ Bm,
                                __bf16* __restrict__ WB, unsigned O, unsigned D,
                                unsigned KA) {
  unsigned idx = blockIdx.x * blockDim.x + threadIdx.x;
  if (idx >= O * KA) return;
  unsigned o = idx / KA, k = idx % KA;
  float v;
  if (k < D) {
    v = Wb[(size_t)o * D + k];
  } else {
    unsigned er = k - D;
    unsigned e = er >> 4, rr = er & 15;  // R = 16
    v = Bm[((size_t)e * O + o) * 16 + rr];
  }
  WB[idx] = (__bf16)v;
}

__global__ void f32_to_bf16_kernel(const float* __restrict__ src,
                                   __bf16* __restrict__ dst, unsigned n) {
  unsigned i = blockIdx.x * blockDim.x + threadIdx.x;
  if (i < n) dst[i] = (__bf16)src[i];
}

__global__ void conv_x_kernel(const float* __restrict__ x,
                              __bf16* __restrict__ xb, unsigned D, unsigned KA,
                              unsigned total) {
  unsigned i = blockIdx.x * blockDim.x + threadIdx.x;
  if (i >= total) return;
  unsigned n = i / D, k = i % D;
  xb[(size_t)n * KA + k] = (__bf16)x[i];
}

// One block per token: wave e computes logit_e; then softmax + top-2 renorm.
__global__ __launch_bounds__(256) void gate_kernel(
    const float* __restrict__ x, const float* __restrict__ Wg,
    float* __restrict__ w_dense, unsigned D) {
  const unsigned n = blockIdx.x;
  const int e = threadIdx.x >> 5;
  const int lane = threadIdx.x & 31;
  const float* xr = x + (size_t)n * D;
  const float* wr = Wg + (size_t)e * D;
  float s = 0.0f;
  for (unsigned k = lane; k < D; k += 32) s += xr[k] * wr[k];
#pragma unroll
  for (int off = 16; off > 0; off >>= 1) s += __shfl_xor(s, off, 32);
  __shared__ float logits[8];
  __shared__ float wout[8];
  if (lane == 0) logits[e] = s;
  __syncthreads();
  if (threadIdx.x == 0) {
    float mx = logits[0];
    for (int i = 1; i < 8; ++i) mx = fmaxf(mx, logits[i]);
    float p[8], sum = 0.0f;
    for (int i = 0; i < 8; ++i) {
      p[i] = __expf(logits[i] - mx);
      sum += p[i];
    }
    for (int i = 0; i < 8; ++i) p[i] /= sum;
    int i0 = 0;
    float v0 = p[0];
    for (int i = 1; i < 8; ++i)
      if (p[i] > v0) { v0 = p[i]; i0 = i; }
    int i1 = -1;
    float v1 = -1.0f;
    for (int i = 0; i < 8; ++i)
      if (i != i0 && p[i] > v1) { v1 = p[i]; i1 = i; }
    const float t = v0 + v1;
    for (int i = 0; i < 8; ++i) wout[i] = 0.0f;
    wout[i0] = v0 / t;
    wout[i1] = v1 / t;
  }
  __syncthreads();
  if (threadIdx.x < 8) w_dense[(size_t)n * 8 + threadIdx.x] = wout[threadIdx.x];
}

__global__ void hw_kernel(const float* __restrict__ h,
                          const float* __restrict__ wd,
                          __bf16* __restrict__ xb, unsigned D, unsigned KA,
                          unsigned total) {
  unsigned i = blockIdx.x * blockDim.x + threadIdx.x;
  if (i >= total) return;
  unsigned n = i >> 7;  // E*R = 128
  unsigned er = i & 127;
  unsigned e = er >> 4;
  float v = h[i] * wd[(size_t)n * 8 + e] * MOE_SCALING;
  xb[(size_t)n * KA + D + er] = (__bf16)v;
}

// ---------------------------------------------------------------------------
extern "C" void kernel_launch(void* const* d_in, const int* in_sizes, int n_in,
                              void* d_out, int out_size, void* d_ws,
                              size_t ws_size, hipStream_t stream) {
  const float* x      = (const float*)d_in[0];
  const float* W_base = (const float*)d_in[1];
  const float* b_base = (const float*)d_in[2];
  const float* W_gate = (const float*)d_in[3];
  const float* A      = (const float*)d_in[4];
  const float* B      = (const float*)d_in[5];
  (void)n_in; (void)out_size; (void)ws_size;

  const unsigned D = 4096, O = 4096, E = 8, R = 16;
  const unsigned N  = (unsigned)(in_sizes[0] / (int)D);  // 8192 tokens
  const unsigned ER = E * R;                              // 128
  const unsigned KA = D + ER;                             // 4224

  char* ws = (char*)d_ws;
  size_t off = 0;
  auto carve = [&](size_t bytes) -> void* {
    void* p = ws + off;
    off = (off + bytes + 255) & ~(size_t)255;
    return p;
  };
  __bf16* xb = (__bf16*)carve((size_t)N * KA * 2);  // [N, 4224]
  __bf16* WB = (__bf16*)carve((size_t)O * KA * 2);  // [O, 4224]
  __bf16* Af = (__bf16*)carve((size_t)ER * D * 2);  // [128, 4096]
  float*  h  = (float*)carve((size_t)N * ER * 4);   // [N, 128]
  float*  wd = (float*)carve((size_t)N * E * 4);    // [N, 8]

  const int T = 256;

  {  // 1) augmented weight matrix (bf16): [W_base | Bflat]
    unsigned total = O * KA;
    build_wb_kernel<<<(total + T - 1) / T, T, 0, stream>>>(W_base, B, WB, O, D,
                                                           KA);
  }
  {  // 2) A -> bf16, contiguous [E*R, D]
    unsigned total = ER * D;
    f32_to_bf16_kernel<<<(total + T - 1) / T, T, 0, stream>>>(A, Af, total);
  }
  {  // 3) x -> bf16 into xb[:, :D]
    unsigned total = N * D;
    conv_x_kernel<<<(total + T - 1) / T, T, 0, stream>>>(x, xb, D, KA, total);
  }
  // 4) gating: softmax + top-2 renormalized weights
  gate_kernel<<<N, 256, 0, stream>>>(x, W_gate, wd, D);

  {  // 5) LoRA down-proj: h[N,128] = xb[:, :D] @ Af^T  (BM=128, BN=128)
    dim3 grid(ER / 128, N / 128);
    gemm_bf16_wmma_lds<4, 2, 2, 4, false><<<grid, 256, 0, stream>>>(
        xb, (int)KA, Af, (int)D, h, (int)ER, nullptr, (int)D);
  }
  {  // 6) gate-weight + scale, write bf16 into xb[:, D:]
    unsigned total = N * ER;
    hw_kernel<<<(total + T - 1) / T, T, 0, stream>>>(h, wd, xb, D, KA, total);
  }
  {  // 7) fused base + MoE GEMM: out = xb @ WB^T + b_base (BM=128, BN=256)
    dim3 grid(O / 256, N / 128);
    gemm_bf16_wmma_lds<4, 4, 2, 4, true><<<grid, 256, 0, stream>>>(
        xb, (int)KA, WB, (int)KA, (float*)d_out, (int)O, b_base, (int)KA);
  }
}
